// TSAD_58162447122692
// MI455X (gfx1250) — compile-verified, matrix operand order checked by reading, not run on hardware
//
#include <hip/hip_runtime.h>
#include <hip/hip_bf16.h>
#include <math.h>

// ---------------------------------------------------------------------------
// Problem constants (from reference)
// ---------------------------------------------------------------------------
static constexpr int Bb   = 8;
static constexpr int Hh   = 64;
static constexpr int Ww   = 64;
static constexpr int CVd  = 256;
static constexpr int Tt   = 77;
static constexpr int CTd  = 512;
static constexpr int NHn  = 8;
static constexpr int DHd  = 32;          // CV / NH
static constexpr int Nv   = Hh * Ww;     // 4096
static constexpr int MROWS = Bb * Nv;    // 32768 visual tokens
static constexpr int TROWS = Bb * Tt;    // 616 text tokens
static constexpr int KMAX  = 1024;       // largest K used (FFN2)

#define NEG_SENT (-3.0e38f)

typedef __attribute__((ext_vector_type(16))) __bf16 bf16x16;
typedef __attribute__((ext_vector_type(8)))  __bf16 bf16x8;
typedef __attribute__((ext_vector_type(8)))  float  f32x8;
typedef __attribute__((ext_vector_type(4)))  unsigned int u32x4;
typedef __attribute__((ext_vector_type(8)))  int i32x8;
typedef __attribute__((ext_vector_type(4)))  int i32x4;

#if defined(__has_builtin)
#  if __has_builtin(__builtin_amdgcn_tensor_load_to_lds)
#    define HAVE_TDM 1
#  else
#    define HAVE_TDM 0
#  endif
#  if __has_builtin(__builtin_amdgcn_s_wait_tensorcnt)
#    define WAIT_TENSORCNT0() __builtin_amdgcn_s_wait_tensorcnt(0)
#  else
#    define WAIT_TENSORCNT0() asm volatile("s_wait_tensorcnt 0x0" ::: "memory")
#  endif
#else
#  define HAVE_TDM 0
#  define WAIT_TENSORCNT0()
#endif

typedef __attribute__((address_space(3))) const __bf16* lds_cptr;

// ---------------------------------------------------------------------------
// WMMA bf16 GEMM:  C[M,N] = A[M,K] @ B[K,N] (+bias, +addsrc, gelu?)
//   A  : bf16 row-major (M,K);  Bt : bf16 row-major (N,K) == B transposed.
// Block = 4 waves. All waves share one 16-row A slab (16 x K bf16, <=32KB),
// DMA'd into LDS once by the Tensor Data Mover (D# built per ISA 08 §8.3/8.4),
// then each wave computes a 16x64 output strip (4 accumulators) so the
// LDS-resident A fragment is reused 4x per K-step. B panels stream from
// global via clause-grouped b128 loads (L2-resident weights).
// Fragment lane layout (16-bit A 16x32, ISA 7.12.2):
//   lane<16 : row=lane,    K in {k0..k0+7} U {k0+16..k0+23}
//   lane>=16: row=lane-16, K in {k0+8..k0+15} U {k0+24..k0+31}
// Each K-group-of-8 is 16 contiguous bytes -> one b128 (ds / global) load.
// ---------------------------------------------------------------------------
__global__ void gemm_bf16_wmma(const __bf16* __restrict__ A,
                               const __bf16* __restrict__ Bt,
                               const float*  __restrict__ bias,
                               const float*  __restrict__ addsrc,
                               float*        __restrict__ outF,
                               __bf16*       __restrict__ outBF,
                               int M, int N, int K, int doGelu)
{
    __shared__ __bf16 As[16 * KMAX];                 // 32 KB A slab

    const int lane  = threadIdx.x;                   // 0..31
    const int wave  = threadIdx.y;                   // 0..3
    const int mTile = blockIdx.y;
    const int nTileBase = blockIdx.x * 16 + wave * 4;  // block covers 256 cols

    // ---- stage A slab (rows mTile*16 .. +15, all K) into LDS --------------
#if HAVE_TDM
    if (wave == 0) {
        const unsigned long long ga =
            (unsigned long long)(const void*)(A + (size_t)mTile * 16 * K);
        const unsigned ldsOff = (unsigned)(unsigned long long)(lds_cptr)&As[0];
        const unsigned rowsRem = (unsigned)(M - mTile * 16);   // OOB rows read 0

        u32x4 g0;
        g0[0] = 1u;                                          // count=1 (valid)
        g0[1] = ldsOff;                                      // lds_addr
        g0[2] = (unsigned)(ga & 0xFFFFFFFFull);              // global_addr lo
        g0[3] = (unsigned)((ga >> 32) & 0x01FFFFFFull)       // global_addr hi
              | (2u << 30);                                  // type = 2 (image)
        i32x8 g1;
        g1[0] = (int)(1u << 16);                             // data_size = 2B
        g1[1] = (int)(((unsigned)K & 0xFFFFu) << 16);        // tensor_dim0 lo
        g1[2] = (int)((((unsigned)K >> 16) & 0xFFFFu)        // tensor_dim0 hi
              | ((rowsRem & 0xFFFFu) << 16));                // tensor_dim1 lo
        g1[3] = (int)(((rowsRem >> 16) & 0xFFFFu)            // tensor_dim1 hi
              | (((unsigned)K & 0xFFFFu) << 16));            // tile_dim0 = K
        g1[4] = 16;                                          // tile_dim1 = 16
        g1[5] = K;                                           // dim0_stride lo
        g1[6] = 0;
        g1[7] = 0;
        i32x4 gz = {0, 0, 0, 0};
#if __clang_major__ >= 23
        i32x8 gz8 = {0, 0, 0, 0, 0, 0, 0, 0};
        __builtin_amdgcn_tensor_load_to_lds(g0, g1, gz, gz, gz8, 0);
#else
        __builtin_amdgcn_tensor_load_to_lds(g0, g1, gz, gz, 0);
#endif
        WAIT_TENSORCNT0();
    }
#else
    {   // cooperative fallback: slab is contiguous (rows are adjacent in memory)
        const int tid = wave * 32 + lane;
        const __bf16* src = A + (size_t)mTile * 16 * K;
        for (int e = tid * 8; e < 16 * K; e += 128 * 8)
            *(bf16x8*)(As + e) = *(const bf16x8*)(src + e);
    }
#endif
    __syncthreads();

    const int half = lane >> 4;              // 0 or 1
    const int lrow = lane & 15;

    // B rows for the 4 N-subtiles of this wave (clamped if beyond N)
    const __bf16* bp[4];
#pragma unroll
    for (int j = 0; j < 4; ++j) {
        int bt = (nTileBase + j) * 16;
        if (bt >= N) bt = 0;
        bp[j] = Bt + (size_t)(bt + lrow) * K;
    }
    const __bf16* arow = As + (size_t)lrow * K;    // LDS, generic pointer

    union Frag { bf16x16 v; bf16x8 h[2]; };

    f32x8 c0 = {}, c1 = {}, c2 = {}, c3 = {};
    for (int k0 = 0; k0 < K; k0 += 32) {
        Frag fa;
        fa.h[0] = *(const bf16x8*)(arow + k0 +      half * 8);   // ds_load_b128
        fa.h[1] = *(const bf16x8*)(arow + k0 + 16 + half * 8);
        Frag fb0, fb1, fb2, fb3;
        fb0.h[0] = *(const bf16x8*)(bp[0] + k0 +      half * 8);
        fb0.h[1] = *(const bf16x8*)(bp[0] + k0 + 16 + half * 8);
        fb1.h[0] = *(const bf16x8*)(bp[1] + k0 +      half * 8);
        fb1.h[1] = *(const bf16x8*)(bp[1] + k0 + 16 + half * 8);
        fb2.h[0] = *(const bf16x8*)(bp[2] + k0 +      half * 8);
        fb2.h[1] = *(const bf16x8*)(bp[2] + k0 + 16 + half * 8);
        fb3.h[0] = *(const bf16x8*)(bp[3] + k0 +      half * 8);
        fb3.h[1] = *(const bf16x8*)(bp[3] + k0 + 16 + half * 8);
        c0 = __builtin_amdgcn_wmma_f32_16x16x32_bf16(false, fa.v, false, fb0.v,
                                                     (short)0, c0, false, false);
        c1 = __builtin_amdgcn_wmma_f32_16x16x32_bf16(false, fa.v, false, fb1.v,
                                                     (short)0, c1, false, false);
        c2 = __builtin_amdgcn_wmma_f32_16x16x32_bf16(false, fa.v, false, fb2.v,
                                                     (short)0, c2, false, false);
        c3 = __builtin_amdgcn_wmma_f32_16x16x32_bf16(false, fa.v, false, fb3.v,
                                                     (short)0, c3, false, false);
    }

    // C/D layout: VGPR r, lane<16 -> (M=r, N=lane); lane>=16 -> (M=8+r, N=lane-16)
    f32x8 cc[4] = {c0, c1, c2, c3};
#pragma unroll
    for (int j = 0; j < 4; ++j) {
        const int col = (nTileBase + j) * 16 + lrow;
        if (col >= N) continue;
        const float bc = bias ? bias[col] : 0.0f;
#pragma unroll
        for (int r = 0; r < 8; ++r) {
            const int row = mTile * 16 + r + half * 8;
            if (row < M) {
                float v = cc[j][r] + bc;
                if (doGelu) v = 0.5f * v * (1.0f + erff(v * 0.7071067811865475f));
                const size_t idx = (size_t)row * N + col;
                if (addsrc) v += addsrc[idx];
                if (outF)  outF[idx]  = v;
                if (outBF) outBF[idx] = (__bf16)v;
            }
        }
    }
}

// ---------------------------------------------------------------------------
// Weight convert + transpose:  in (K,N) f32  ->  out (N,K) bf16
// ---------------------------------------------------------------------------
__global__ void cvt_transpose(const float* __restrict__ in, __bf16* __restrict__ out,
                              int K, int N)
{
    int idx = blockIdx.x * blockDim.x + threadIdx.x;
    if (idx >= K * N) return;
    int k = idx / N, n = idx - k * N;
    out[(size_t)n * K + k] = (__bf16)in[idx];
}

// Elementwise f32 -> bf16
__global__ void cvt_bf16(const float* __restrict__ in, __bf16* __restrict__ out, int n)
{
    int i = blockIdx.x * blockDim.x + threadIdx.x;
    if (i < n) out[i] = (__bf16)in[i];
}

// ---------------------------------------------------------------------------
// LayerNorm over last dim (256), dual f32 + bf16 output. One thread per row.
// ---------------------------------------------------------------------------
__global__ void ln_fwd(const float* __restrict__ in,
                       const float* __restrict__ g, const float* __restrict__ b,
                       float* __restrict__ outF, __bf16* __restrict__ outBF, int rows)
{
    int r = blockIdx.x * blockDim.x + threadIdx.x;
    if (r >= rows) return;
    const float* x = in + (size_t)r * CVd;
    float mu = 0.f;
    for (int i = 0; i < CVd; ++i) mu += x[i];
    mu *= (1.0f / CVd);
    float var = 0.f;
    for (int i = 0; i < CVd; ++i) { float d = x[i] - mu; var += d * d; }
    var *= (1.0f / CVd);
    float inv = rsqrtf(var + 1e-5f);
    for (int i = 0; i < CVd; ++i) {
        float v = (x[i] - mu) * inv * g[i] + b[i];
        outF[(size_t)r * CVd + i]  = v;
        outBF[(size_t)r * CVd + i] = (__bf16)v;
    }
}

// residual (y = x + alpha*wo) then LayerNorm(g2,b2); y2 written f32+bf16.
// outF may alias x (each thread owns its row).
__global__ void resid_ln(const float* __restrict__ x, const float* __restrict__ wo,
                         const float* __restrict__ alpha_p,
                         const float* __restrict__ g, const float* __restrict__ b,
                         float* __restrict__ outF, __bf16* __restrict__ outBF, int rows)
{
    int r = blockIdx.x * blockDim.x + threadIdx.x;
    if (r >= rows) return;
    const float a = alpha_p[0];
    const float* xr = x  + (size_t)r * CVd;
    const float* wr = wo + (size_t)r * CVd;
    float mu = 0.f;
    for (int i = 0; i < CVd; ++i) mu += xr[i] + a * wr[i];
    mu *= (1.0f / CVd);
    float var = 0.f;
    for (int i = 0; i < CVd; ++i) { float d = xr[i] + a * wr[i] - mu; var += d * d; }
    var *= (1.0f / CVd);
    float inv = rsqrtf(var + 1e-5f);
    for (int i = 0; i < CVd; ++i) {
        float v = (xr[i] + a * wr[i] - mu) * inv * g[i] + b[i];
        outF[(size_t)r * CVd + i]  = v;
        outBF[(size_t)r * CVd + i] = (__bf16)v;
    }
}

// ---------------------------------------------------------------------------
// Pad detection: pad[b,t] = sum|text[b,t,:]| <= 1e-6 ; all_pad[b] = AND_t pad
// ---------------------------------------------------------------------------
__global__ void pad_kernel(const float* __restrict__ text, int* __restrict__ pad, int bt)
{
    int i = blockIdx.x * blockDim.x + threadIdx.x;
    if (i >= bt) return;
    const float* row = text + (size_t)i * CTd;
    float s = 0.f;
    for (int c = 0; c < CTd; ++c) s += fabsf(row[c]);
    pad[i] = (s <= 1e-6f) ? 1 : 0;
}

__global__ void allpad_kernel(const int* __restrict__ pad, int* __restrict__ allpad)
{
    int b = blockIdx.x * blockDim.x + threadIdx.x;
    if (b >= Bb) return;
    int all = 1;
    for (int t = 0; t < Tt; ++t) all &= pad[b * Tt + t];
    allpad[b] = all;
}

// Per-(b,t,head) L2-normalize K rows (k: (TROWS,256) f32 -> kn f32)
__global__ void knorm_kernel(const float* __restrict__ k, float* __restrict__ kn)
{
    int i = blockIdx.x * blockDim.x + threadIdx.x;      // (b*T)*NH + h
    if (i >= TROWS * NHn) return;
    int bt = i / NHn, h = i - bt * NHn;
    const float* src = k + (size_t)bt * CVd + h * DHd;
    float*       dst = kn + (size_t)bt * CVd + h * DHd;
    float s = 0.f;
    for (int d = 0; d < DHd; ++d) s += src[d] * src[d];
    float inv = 1.0f / fmaxf(sqrtf(s), 1e-6f);
    for (int d = 0; d < DHd; ++d) dst[d] = src[d] * inv;
}

// ---------------------------------------------------------------------------
// Cosine attention, top-5 sparsified softmax, attn @ V.
// One thread per query row; sim row staged in LDS. Block = 64 threads.
// ---------------------------------------------------------------------------
__global__ void attn_kernel(const float* __restrict__ q,   // (MROWS,256) unnormalized
                            const float* __restrict__ kn,  // (TROWS,256) head-normalized
                            const float* __restrict__ v,   // (TROWS,256)
                            const int*   __restrict__ pad,
                            const int*   __restrict__ allpad,
                            const float* __restrict__ logit_scale_p,
                            __bf16*      __restrict__ alignedBF)
{
    __shared__ float sim[64][80];
    const int gid = blockIdx.x * 64 + threadIdx.x;    // query row in [0, MROWS)
    const int b   = gid >> 12;                        // Nv == 4096
    float ls = logit_scale_p[0];
    ls = fminf(fmaxf(ls, -2.0f), 2.0f);
    const float scale = __expf(ls) * 0.17677669529663687f;   // 1/sqrt(32)
    const bool ap = (allpad[b] != 0);
    const float* qrow = q + (size_t)gid * CVd;
    float* srow = sim[threadIdx.x];

    for (int h = 0; h < NHn; ++h) {
        float qh[DHd];
        float nq = 0.f;
#pragma unroll
        for (int d = 0; d < DHd; ++d) { qh[d] = qrow[h * DHd + d]; nq += qh[d] * qh[d]; }
        const float invq = 1.0f / fmaxf(sqrtf(nq), 1e-6f);

        for (int t = 0; t < Tt; ++t) {
            if (pad[b * Tt + t]) { srow[t] = NEG_SENT; continue; }
            const float* kr = kn + (size_t)(b * Tt + t) * CVd + h * DHd;
            float s = 0.f;
#pragma unroll
            for (int d = 0; d < DHd; ++d) s += qh[d] * kr[d];
            srow[t] = s * invq * scale;
        }

        // top-5 selection (first-occurrence argmax, matching top_k tie order)
        float wv[5]; int wi[5];
        for (int j = 0; j < 5; ++j) {
            float best = NEG_SENT; int bi = 0;
            for (int t = 0; t < Tt; ++t)
                if (srow[t] > best) { best = srow[t]; bi = t; }
            wv[j] = best; wi[j] = bi;
            srow[bi] = NEG_SENT;
        }

        float acc[DHd];
#pragma unroll
        for (int d = 0; d < DHd; ++d) acc[d] = 0.f;

        const float m = wv[0];
        if (!ap && m > -1.0e37f) {
            float e[5]; float den = 0.f;
            for (int j = 0; j < 5; ++j) {
                e[j] = (wv[j] > -1.0e37f) ? __expf(wv[j] - m) : 0.f;
                den += e[j];
            }
            const float invden = 1.0f / den;
            for (int j = 0; j < 5; ++j) {
                if (e[j] > 0.f) {
                    const float w = e[j] * invden;
                    const float* vr = v + (size_t)(b * Tt + wi[j]) * CVd + h * DHd;
#pragma unroll
                    for (int d = 0; d < DHd; ++d) acc[d] += w * vr[d];
                }
            }
        }
#pragma unroll
        for (int d = 0; d < DHd; ++d)
            alignedBF[(size_t)gid * CVd + h * DHd + d] = (__bf16)acc[d];
    }
}

// ---------------------------------------------------------------------------
// Host launch
// ---------------------------------------------------------------------------
extern "C" void kernel_launch(void* const* d_in, const int* in_sizes, int n_in,
                              void* d_out, int out_size, void* d_ws, size_t ws_size,
                              hipStream_t stream)
{
    (void)in_sizes; (void)n_in; (void)out_size; (void)ws_size;

    const float* visual = (const float*)d_in[0];
    const float* text   = (const float*)d_in[1];
    const float* Wq = (const float*)d_in[2];  const float* bq = (const float*)d_in[3];
    const float* Wk = (const float*)d_in[4];  const float* bk = (const float*)d_in[5];
    const float* Wv = (const float*)d_in[6];  const float* bv = (const float*)d_in[7];
    const float* Wo = (const float*)d_in[8];  const float* bo = (const float*)d_in[9];
    const float* g1 = (const float*)d_in[10]; const float* b1 = (const float*)d_in[11];
    const float* g2 = (const float*)d_in[12]; const float* b2 = (const float*)d_in[13];
    const float* W1 = (const float*)d_in[14]; const float* bf1 = (const float*)d_in[15];
    const float* W2 = (const float*)d_in[16]; const float* bf2 = (const float*)d_in[17];
    const float* logit_scale = (const float*)d_in[18];
    const float* alpha       = (const float*)d_in[19];
    float* out = (float*)d_out;

    // workspace bump allocator
    char* ws = (char*)d_ws;
    size_t off = 0;
    auto alloc = [&](size_t bytes) -> char* {
        char* p = ws + off;
        off += (bytes + 255) & ~(size_t)255;
        return p;
    };
    float*  x_f32   = (float*) alloc((size_t)MROWS * CVd * 4);   // also y2 (in-place)
    __bf16* x_bf    = (__bf16*)alloc((size_t)MROWS * CVd * 2);   // also y2_bf
    float*  q_f32   = (float*) alloc((size_t)MROWS * CVd * 4);   // also Wo output
    __bf16* algn_bf = (__bf16*)alloc((size_t)MROWS * CVd * 2);
    __bf16* h_bf    = (__bf16*)alloc((size_t)MROWS * 4 * CVd * 2);
    __bf16* text_bf = (__bf16*)alloc((size_t)TROWS * CTd * 2);
    float*  k_f32   = (float*) alloc((size_t)TROWS * CVd * 4);
    float*  kn_f32  = (float*) alloc((size_t)TROWS * CVd * 4);
    float*  v_f32   = (float*) alloc((size_t)TROWS * CVd * 4);
    __bf16* Wq_t    = (__bf16*)alloc((size_t)CVd * CVd * 2);
    __bf16* Wk_t    = (__bf16*)alloc((size_t)CTd * CVd * 2);
    __bf16* Wv_t    = (__bf16*)alloc((size_t)CTd * CVd * 2);
    __bf16* Wo_t    = (__bf16*)alloc((size_t)CVd * CVd * 2);
    __bf16* W1_t    = (__bf16*)alloc((size_t)CVd * 4 * CVd * 2);
    __bf16* W2_t    = (__bf16*)alloc((size_t)4 * CVd * CVd * 2);
    int*    pad     = (int*)   alloc((size_t)TROWS * 4);
    int*    allpad  = (int*)   alloc((size_t)Bb * 4);

    auto cdiv = [](int a, int b) { return (a + b - 1) / b; };

    // 1. weight convert+transpose (f32 -> bf16, N-major)
    {
        int n;
        n = CVd * CVd;       cvt_transpose<<<cdiv(n,256),256,0,stream>>>(Wq, Wq_t, CVd, CVd);
        n = CTd * CVd;       cvt_transpose<<<cdiv(n,256),256,0,stream>>>(Wk, Wk_t, CTd, CVd);
        n = CTd * CVd;       cvt_transpose<<<cdiv(n,256),256,0,stream>>>(Wv, Wv_t, CTd, CVd);
        n = CVd * CVd;       cvt_transpose<<<cdiv(n,256),256,0,stream>>>(Wo, Wo_t, CVd, CVd);
        n = CVd * 4 * CVd;   cvt_transpose<<<cdiv(n,256),256,0,stream>>>(W1, W1_t, CVd, 4*CVd);
        n = 4 * CVd * CVd;   cvt_transpose<<<cdiv(n,256),256,0,stream>>>(W2, W2_t, 4*CVd, CVd);
    }
    // 2. text -> bf16 ; pad masks
    {
        int n = TROWS * CTd;
        cvt_bf16<<<cdiv(n,256),256,0,stream>>>(text, text_bf, n);
        pad_kernel<<<cdiv(TROWS,128),128,0,stream>>>(text, pad, TROWS);
        allpad_kernel<<<1,32,0,stream>>>(pad, allpad);
    }
    // 3. pre-LN on visual features -> x (f32 + bf16)
    ln_fwd<<<cdiv(MROWS,128),128,0,stream>>>(visual, g1, b1, x_f32, x_bf, MROWS);

    // 4. projections (WMMA); each block = 16 rows x 256 cols
    {
        dim3 blk(32, 4);
        gemm_bf16_wmma<<<dim3(cdiv(CVd,256), cdiv(MROWS,16)), blk, 0, stream>>>(
            x_bf, Wq_t, bq, nullptr, q_f32, nullptr, MROWS, CVd, CVd, 0);
        gemm_bf16_wmma<<<dim3(cdiv(CVd,256), cdiv(TROWS,16)), blk, 0, stream>>>(
            text_bf, Wk_t, bk, nullptr, k_f32, nullptr, TROWS, CVd, CTd, 0);
        gemm_bf16_wmma<<<dim3(cdiv(CVd,256), cdiv(TROWS,16)), blk, 0, stream>>>(
            text_bf, Wv_t, bv, nullptr, v_f32, nullptr, TROWS, CVd, CTd, 0);
    }
    // 5. normalize K heads, run attention
    knorm_kernel<<<cdiv(TROWS*NHn,128),128,0,stream>>>(k_f32, kn_f32);
    attn_kernel<<<MROWS/64, 64, 0, stream>>>(q_f32, kn_f32, v_f32, pad, allpad,
                                             logit_scale, algn_bf);
    // 6. output projection (overwrites q_f32 buffer)
    {
        dim3 blk(32, 4);
        gemm_bf16_wmma<<<dim3(cdiv(CVd,256), cdiv(MROWS,16)), blk, 0, stream>>>(
            algn_bf, Wo_t, bo, nullptr, q_f32, nullptr, MROWS, CVd, CVd, 0);
    }
    // 7. residual + LN2 -> y2 (in-place into x buffers)
    resid_ln<<<cdiv(MROWS,128),128,0,stream>>>(x_f32, q_f32, alpha, g2, b2,
                                               x_f32, x_bf, MROWS);
    // 8. FFN1 with fused exact-GELU -> h_bf ; FFN2 fused with residual -> d_out
    {
        dim3 blk(32, 4);
        gemm_bf16_wmma<<<dim3(cdiv(4*CVd,256), cdiv(MROWS,16)), blk, 0, stream>>>(
            x_bf, W1_t, bf1, nullptr, nullptr, h_bf, MROWS, 4*CVd, CVd, 1);
        gemm_bf16_wmma<<<dim3(cdiv(CVd,256), cdiv(MROWS,16)), blk, 0, stream>>>(
            h_bf, W2_t, bf2, x_f32, out, nullptr, MROWS, CVd, 4*CVd, 0);
    }
}
